// VectorQuantizer_30648886624694
// MI455X (gfx1250) — compile-verified, matrix operand order checked by reading, not run on hardware
//
#include <hip/hip_runtime.h>

// ---------------- constants (match reference) ----------------
#define Bsz   1024
#define Lt    64
#define Gg    32
#define Dm    512
#define Hh    8
#define HDim  64
#define NEcb  32768
#define KTOP  5
#define S0n   768
#define S1n   256
#define REGION 10922   // NE // 3

// ---------------- bf16 helpers (storage = ushort) ----------------
typedef __attribute__((ext_vector_type(16))) __bf16 v16bf;
typedef __attribute__((ext_vector_type(8)))  float  v8f;

union FragU { uint4 q[2]; unsigned int u[8]; v16bf v; };

__device__ __forceinline__ unsigned short f2bf(float f) {
  unsigned int u = __builtin_bit_cast(unsigned int, f);
  unsigned int r = u + 0x7FFFu + ((u >> 16) & 1u);   // RNE
  return (unsigned short)(r >> 16);
}
__device__ __forceinline__ float bf2f(unsigned short h) {
  unsigned int u = ((unsigned int)h) << 16;
  return __builtin_bit_cast(float, u);
}

// Fragment loader for v_wmma_f32_16x16x32_bf16, wave32.
// rowptr = start of this lane's matrix row (row m = lane%16 for A, row n =
// lane%16 of W for B; every GEMM here is X @ W^T with W row-major).
// 16-bit 16x32 layout: lanes 0-15 hold K pairs {0..7,16..23}, lanes 16-31
// hold {8..15,24..31}  ==> exactly two contiguous 16-byte chunks per lane.
__device__ __forceinline__ v16bf frag_load(const unsigned short* rowptr, int kk,
                                           int lane) {
  const int base = kk + (((lane >> 4) & 1) << 3);
  FragU f;
  f.q[0] = *(const uint4*)(rowptr + base);        // K pairs 0..3 (this half)
  f.q[1] = *(const uint4*)(rowptr + base + 16);   // K pairs 4..7 (this half)
  return f.v;
}

__device__ __forceinline__ v8f wmma_bf16(v16bf a, v16bf b, v8f c) {
  return __builtin_amdgcn_wmma_f32_16x16x32_bf16(false, a, false, b, (short)0, c,
                                                 false, false);
}

// ---------------- elementwise / reduction kernels ----------------
__global__ void copyconv_kernel(const float* __restrict__ src,
                                float* __restrict__ dstf,
                                unsigned short* __restrict__ dstbf, long n) {
  for (long i = blockIdx.x * (long)blockDim.x + threadIdx.x; i < n;
       i += (long)gridDim.x * blockDim.x) {
    float v = src[i];
    if (dstf) dstf[i] = v;
    dstbf[i] = f2bf(v);
  }
}

// row-wise L2 normalize (D=512) -> bf16
__global__ __launch_bounds__(256) void l2n_kernel(const float* __restrict__ src,
                                                  unsigned short* __restrict__ dst) {
  __shared__ float red[256];
  const int row = blockIdx.x, tid = threadIdx.x;
  const long base = (long)row * Dm;
  float v0 = src[base + tid], v1 = src[base + tid + 256];
  red[tid] = v0 * v0 + v1 * v1;
  __syncthreads();
  for (int s = 128; s > 0; s >>= 1) {
    if (tid < s) red[tid] += red[tid + s];
    __syncthreads();
  }
  float nrm = sqrtf(red[0]);
  nrm = fmaxf(nrm, 1e-12f);
  const float inv = 1.0f / nrm;
  dst[base + tid] = f2bf(v0 * inv);
  dst[base + tid + 256] = f2bf(v1 * inv);
}

// x = LN(x + ao)*g + b ; also emit bf16 copy
__global__ __launch_bounds__(256) void resln_kernel(
    float* __restrict__ x, unsigned short* __restrict__ xbf,
    const float* __restrict__ ao, const float* __restrict__ gam,
    const float* __restrict__ bet) {
  __shared__ float r1[256], r2[256];
  const int row = blockIdx.x, tid = threadIdx.x;
  const long base = (long)row * Dm;
  float v0 = x[base + tid] + ao[base + tid];
  float v1 = x[base + tid + 256] + ao[base + tid + 256];
  r1[tid] = v0 + v1;
  r2[tid] = v0 * v0 + v1 * v1;
  __syncthreads();
  for (int s = 128; s > 0; s >>= 1) {
    if (tid < s) { r1[tid] += r1[tid + s]; r2[tid] += r2[tid + s]; }
    __syncthreads();
  }
  const float mean = r1[0] * (1.0f / Dm);
  const float var  = r2[0] * (1.0f / Dm) - mean * mean;
  const float inv  = rsqrtf(var + 1e-5f);
  float o0 = (v0 - mean) * inv * gam[tid] + bet[tid];
  float o1 = (v1 - mean) * inv * gam[tid + 256] + bet[tid + 256];
  x[base + tid] = o0;       xbf[base + tid] = f2bf(o0);
  x[base + tid + 256] = o1; xbf[base + tid + 256] = f2bf(o1);
}

__global__ void extract_t0_kernel(const float* __restrict__ x, float* __restrict__ zt) {
  int i = blockIdx.x * blockDim.x + threadIdx.x;
  if (i < Bsz * Dm) {
    int b = i >> 9, d = i & 511;
    zt[i] = x[((long)b * Lt) * Dm + d];
  }
}

__global__ void meang_kernel(const float* __restrict__ x, float* __restrict__ zg) {
  int i = blockIdx.x * blockDim.x + threadIdx.x;
  if (i < Bsz * Dm) {
    int b = i >> 9, d = i & 511;
    float acc = 0.f;
#pragma unroll
    for (int g = 0; g < Gg; ++g) acc += x[((long)(b * Gg + g)) * Dm + d];
    zg[i] = acc * (1.0f / Gg);
  }
}

// ---------------- WMMA GEMM: Y(MxN) = Xbf(MxK) @ Wbf^T(NxK) + bias ----------------
// One wave -> one 16x16 M-tile x four 16-wide N-tiles (A-fragment reuse).
// Edge N-tiles: invalid W rows are CLAMPED to row 0 (loads stay unconditional
// and vectorized); the garbage columns are simply never stored.
__global__ __launch_bounds__(256) void gemm_bf16_kernel(
    const unsigned short* __restrict__ X, int ldx,
    const unsigned short* __restrict__ W, int ldw,
    const float* __restrict__ bias,
    float* __restrict__ Yf, unsigned short* __restrict__ Ybf, int ldy,
    int M, int N, int K) {
  const int lane = threadIdx.x & 31;
  const long wid = (long)blockIdx.x * (blockDim.x >> 5) + (threadIdx.x >> 5);
  const int NT  = (N + 15) >> 4;
  const int NTG = (NT + 3) >> 2;
  const int MT  = M >> 4;
  if (wid >= (long)MT * NTG) return;            // wave-uniform exit (EXEC stays full)
  const int tm  = (int)(wid / NTG);
  const int tng = (int)(wid % NTG);

  const unsigned short* xrow = X + (long)(tm * 16 + (lane & 15)) * ldx;

  int nr0 = (tng * 4 + 0) * 16 + (lane & 15);
  int nr1 = (tng * 4 + 1) * 16 + (lane & 15);
  int nr2 = (tng * 4 + 2) * 16 + (lane & 15);
  int nr3 = (tng * 4 + 3) * 16 + (lane & 15);
  const unsigned short* wrow0 = W + (long)(nr0 < N ? nr0 : 0) * ldw;
  const unsigned short* wrow1 = W + (long)(nr1 < N ? nr1 : 0) * ldw;
  const unsigned short* wrow2 = W + (long)(nr2 < N ? nr2 : 0) * ldw;
  const unsigned short* wrow3 = W + (long)(nr3 < N ? nr3 : 0) * ldw;

  v8f acc0 = {}, acc1 = {}, acc2 = {}, acc3 = {};
  for (int kk = 0; kk < K; kk += 32) {
    __builtin_prefetch(xrow + kk + 64, 0, 3);   // global_prefetch_b8: next K slice
    v16bf a = frag_load(xrow, kk, lane);
    v16bf b0 = frag_load(wrow0, kk, lane);
    acc0 = wmma_bf16(a, b0, acc0);
    v16bf b1 = frag_load(wrow1, kk, lane);
    acc1 = wmma_bf16(a, b1, acc1);
    v16bf b2 = frag_load(wrow2, kk, lane);
    acc2 = wmma_bf16(a, b2, acc2);
    v16bf b3 = frag_load(wrow3, kk, lane);
    acc3 = wmma_bf16(a, b3, acc3);
  }

  const int mbase = tm * 16 + ((lane >> 4) << 3);
#pragma unroll
  for (int j = 0; j < 4; ++j) {
    const v8f& c = (j == 0) ? acc0 : (j == 1) ? acc1 : (j == 2) ? acc2 : acc3;
    int n = (tng * 4 + j) * 16 + (lane & 15);
    if (n < N) {
      float bs = bias ? bias[n] : 0.0f;
#pragma unroll
      for (int r = 0; r < 8; ++r) {
        float val = c[r] + bs;
        long off = (long)(mbase + r) * ldy + n;
        if (Yf) Yf[off] = val;
        if (Ybf) Ybf[off] = f2bf(val);
      }
    }
  }
}

// ---------------- fused attention: one block per (b,h), 4 waves ----------------
__global__ __launch_bounds__(128) void attn_kernel(
    const unsigned short* __restrict__ qbf, const unsigned short* __restrict__ kbf,
    const unsigned short* __restrict__ vbf, const int* __restrict__ mask,
    unsigned short* __restrict__ obf, int Lq, int Lk) {
  __shared__ __align__(16) float sS[2048];            // Lq*Lk <= 2048
  __shared__ __align__(16) unsigned short sP[2048];
  __shared__ __align__(16) unsigned short sVt[4096];  // HD x Lk (V transposed)
  const int tid = threadIdx.x;
  const int lane = tid & 31;
  const int wid = tid >> 5;
  const int b = blockIdx.x >> 3;          // H = 8
  const int h = blockIdx.x & 7;

  // load V^T into LDS (coalesced over n)
  for (int idx = tid; idx < HDim * Lk; idx += 128) {
    const int n = idx & (HDim - 1);
    const int k = idx >> 6;
    sVt[n * Lk + k] = vbf[((long)(b * Lk + k)) * Dm + h * HDim + n];
  }
  __syncthreads();

  // scores S = scale * Qh @ Kh^T
  const int qt = Lq >> 4, kt = Lk >> 4;
  for (int t = wid; t < qt * kt; t += 4) {
    const int tm = t / kt, tn = t % kt;
    const unsigned short* qrow =
        qbf + ((long)(b * Lq + tm * 16 + (lane & 15))) * Dm + h * HDim;
    const unsigned short* krow =
        kbf + ((long)(b * Lk + tn * 16 + (lane & 15))) * Dm + h * HDim;
    v8f c = {};
#pragma unroll
    for (int kk = 0; kk < HDim; kk += 32) {
      v16bf a = frag_load(qrow, kk, lane);
      v16bf bb = frag_load(krow, kk, lane);
      c = wmma_bf16(a, bb, c);
    }
    const int n = tn * 16 + (lane & 15);
    const int mb = tm * 16 + ((lane >> 4) << 3);
#pragma unroll
    for (int r = 0; r < 8; ++r) sS[(mb + r) * Lk + n] = c[r] * 0.125f;  // 1/sqrt(64)
  }
  __syncthreads();

  // masked softmax along keys
  for (int row = tid; row < Lq; row += 128) {
    float mx = -1e30f;
    for (int j = 0; j < Lk; ++j) {
      float s = sS[row * Lk + j];
      if (mask && mask[b * Lk + j] == 0) s = -1.0e9f;
      sS[row * Lk + j] = s;
      mx = fmaxf(mx, s);
    }
    float sum = 0.f;
    for (int j = 0; j < Lk; ++j) {
      float e = expf(sS[row * Lk + j] - mx);
      sS[row * Lk + j] = e;
      sum += e;
    }
    const float inv = 1.0f / sum;
    for (int j = 0; j < Lk; ++j) sP[row * Lk + j] = f2bf(sS[row * Lk + j] * inv);
  }
  __syncthreads();

  // O = P @ V   (W := V^T, row-major in LDS -> ds_load_b128 fragments)
  const int nt = HDim >> 4;
  for (int t = wid; t < qt * nt; t += 4) {
    const int tm = t / nt, tn = t % nt;
    const unsigned short* prow = sP + (tm * 16 + (lane & 15)) * Lk;
    const unsigned short* wrow = sVt + (tn * 16 + (lane & 15)) * Lk;
    v8f c = {};
    for (int kk = 0; kk < Lk; kk += 32) {
      v16bf a = frag_load(prow, kk, lane);
      v16bf bb = frag_load(wrow, kk, lane);
      c = wmma_bf16(a, bb, c);
    }
    const int n = tn * 16 + (lane & 15);
    const int mb = tm * 16 + ((lane >> 4) << 3);
#pragma unroll
    for (int r = 0; r < 8; ++r)
      obf[((long)(b * Lq + mb + r)) * Dm + h * HDim + n] = f2bf(c[r]);
  }
}

// ---------------- VQ top-K + softmax(2s) + codebook reconstruction ----------------
__global__ __launch_bounds__(256) void vq_topk_kernel(
    const float* __restrict__ S, int N, const unsigned short* __restrict__ cbn,
    float* __restrict__ out, int sliceOff) {
  __shared__ float cv[256 * KTOP];
  __shared__ int   ci[256 * KTOP];
  __shared__ float redv[256];
  __shared__ int   redi[256];
  __shared__ float topv[KTOP];
  __shared__ int   topi[KTOP];
  __shared__ float wgt[KTOP];
  const int b = blockIdx.x, tid = threadIdx.x;
  const float* row = S + (long)b * N;

  float tv[KTOP]; int ti[KTOP];
#pragma unroll
  for (int k = 0; k < KTOP; ++k) { tv[k] = -1e30f; ti[k] = 0; }
  for (int j = tid; j < N; j += 256) {
    float v = row[j];
    if (v > tv[KTOP - 1]) {
      int p = KTOP - 1;
      while (p > 0 && v > tv[p - 1]) { tv[p] = tv[p - 1]; ti[p] = ti[p - 1]; --p; }
      tv[p] = v; ti[p] = j;
    }
  }
#pragma unroll
  for (int k = 0; k < KTOP; ++k) { cv[tid * KTOP + k] = tv[k]; ci[tid * KTOP + k] = ti[k]; }
  __syncthreads();

  for (int r = 0; r < KTOP; ++r) {
    float mv = -1e30f; int ms = 0;
#pragma unroll
    for (int s = 0; s < KTOP; ++s) {
      float v = cv[tid * KTOP + s];
      if (v > mv) { mv = v; ms = tid * KTOP + s; }
    }
    redv[tid] = mv; redi[tid] = ms;
    __syncthreads();
    for (int s = 128; s > 0; s >>= 1) {
      if (tid < s && redv[tid + s] > redv[tid]) {
        redv[tid] = redv[tid + s]; redi[tid] = redi[tid + s];
      }
      __syncthreads();
    }
    if (tid == 0) {
      topv[r] = redv[0];
      topi[r] = ci[redi[0]];
      cv[redi[0]] = -1e31f;
    }
    __syncthreads();
  }
  if (tid == 0) {
    float mx = topv[0], sum = 0.f, e[KTOP];
#pragma unroll
    for (int k = 0; k < KTOP; ++k) { e[k] = expf(2.0f * (topv[k] - mx)); sum += e[k]; }
#pragma unroll
    for (int k = 0; k < KTOP; ++k) wgt[k] = e[k] / sum;
  }
  __syncthreads();
  for (int d = tid; d < Dm; d += 256) {
    float acc = 0.f;
#pragma unroll
    for (int k = 0; k < KTOP; ++k)
      acc += wgt[k] * bf2f(cbn[(long)topi[k] * Dm + d]);
    out[(long)b * 2048 + sliceOff + d] = acc;
  }
}

// ---------------- host orchestration ----------------
extern "C" void kernel_launch(void* const* d_in, const int* in_sizes, int n_in,
                              void* d_out, int out_size, void* d_ws, size_t ws_size,
                              hipStream_t stream) {
  (void)in_sizes; (void)n_in; (void)out_size; (void)ws_size;
  const float* z    = (const float*)d_in[0];
  const float* tf   = (const float*)d_in[1];
  const float* gf   = (const float*)d_in[2];
  const int*   msk  = (const int*)d_in[3];
  const float* cb   = (const float*)d_in[5];
  const float* ptw  = (const float*)d_in[6];
  const float* ptb  = (const float*)d_in[7];
  const float* pgw  = (const float*)d_in[8];
  const float* pgb  = (const float*)d_in[9];
  const float* wi   = (const float*)d_in[10];
  const float* bi   = (const float*)d_in[11];
  const float* wo   = (const float*)d_in[12];
  const float* bo   = (const float*)d_in[13];
  const float* lng  = (const float*)d_in[14];
  const float* lnb  = (const float*)d_in[15];
  float* out = (float*)d_out;

  char* wsp = (char*)d_ws;
  size_t off = 0;
  auto alloc = [&](size_t bytes) -> void* {
    void* p = wsp + off;
    off += (bytes + 255) & ~(size_t)255;
    return p;
  };
  const size_t BIG_F  = (size_t)Bsz * Lt * Dm * 4;  // 128 MB (== B*NE*4 score buffer)
  const size_t BIG_BF = (size_t)Bsz * Lt * Dm * 2;  // 64 MB

  float* Af            = (float*)alloc(BIG_F);   // residual stream x / VQ scores
  float* Hf            = (float*)alloc(BIG_F);   // out-projection f32
  unsigned short* Xbf  = (unsigned short*)alloc(BIG_BF);
  unsigned short* KVbf = (unsigned short*)alloc(BIG_BF);
  unsigned short* Qbf  = (unsigned short*)alloc(BIG_BF);
  unsigned short* Kbf  = (unsigned short*)alloc(BIG_BF);
  unsigned short* Vbf  = (unsigned short*)alloc(BIG_BF);
  unsigned short* Obf  = (unsigned short*)alloc(BIG_BF);
  unsigned short* cbn  = (unsigned short*)alloc((size_t)NEcb * Dm * 2);
  unsigned short* wiB  = (unsigned short*)alloc((size_t)6 * Dm * Dm * 2);
  unsigned short* woB  = (unsigned short*)alloc((size_t)2 * Dm * Dm * 2);
  unsigned short* ptwB = (unsigned short*)alloc((size_t)Dm * S0n * 2);
  unsigned short* pgwB = (unsigned short*)alloc((size_t)Dm * S1n * 2);
  unsigned short* zB   = (unsigned short*)alloc((size_t)Bsz * 1024 * 2);
  float* ztF           = (float*)alloc((size_t)Bsz * Dm * 4);
  float* zgF           = (float*)alloc((size_t)Bsz * Dm * 4);
  float* ptF           = (float*)alloc((size_t)Bsz * Dm * 4);
  float* pgF           = (float*)alloc((size_t)Bsz * Dm * 4);
  unsigned short* ztN  = (unsigned short*)alloc((size_t)Bsz * Dm * 2);
  unsigned short* zgN  = (unsigned short*)alloc((size_t)Bsz * Dm * 2);
  unsigned short* ptN  = (unsigned short*)alloc((size_t)Bsz * Dm * 2);
  unsigned short* pgN  = (unsigned short*)alloc((size_t)Bsz * Dm * 2);

  auto gemm = [&](const unsigned short* X, int ldx, const unsigned short* W, int ldw,
                  const float* bias, float* Yf, unsigned short* Ybf, int ldy,
                  int M, int N, int K) {
    int NT = (N + 15) >> 4;
    int NTG = (NT + 3) >> 2;
    long waves = (long)(M >> 4) * NTG;
    int blocks = (int)((waves + 7) / 8);
    gemm_bf16_kernel<<<blocks, 256, 0, stream>>>(X, ldx, W, ldw, bias, Yf, Ybf, ldy,
                                                 M, N, K);
  };

  // ---- prep: normalized codebook + bf16 weights ----
  l2n_kernel<<<NEcb, 256, 0, stream>>>(cb, cbn);
  copyconv_kernel<<<4096, 256, 0, stream>>>(wi, nullptr, wiB, (long)6 * Dm * Dm);
  copyconv_kernel<<<2048, 256, 0, stream>>>(wo, nullptr, woB, (long)2 * Dm * Dm);
  copyconv_kernel<<<512, 256, 0, stream>>>(ptw, nullptr, ptwB, (long)Dm * S0n);
  copyconv_kernel<<<512, 256, 0, stream>>>(pgw, nullptr, pgwB, (long)Dm * S1n);
  copyconv_kernel<<<1024, 256, 0, stream>>>(z, nullptr, zB, (long)Bsz * 1024);

  // ---- t path: text attends to graph ----
  copyconv_kernel<<<8192, 256, 0, stream>>>(tf, Af, Xbf, (long)Bsz * Lt * Dm);
  copyconv_kernel<<<8192, 256, 0, stream>>>(gf, nullptr, KVbf, (long)Bsz * Gg * Dm);
  for (int l = 0; l < 2; ++l) {
    const unsigned short* wq = wiB + (long)(l * 3 + 0) * Dm * Dm;
    const unsigned short* wk = wiB + (long)(l * 3 + 1) * Dm * Dm;
    const unsigned short* wv = wiB + (long)(l * 3 + 2) * Dm * Dm;
    gemm(Xbf, Dm, wq, Dm, bi + (l * 3 + 0) * Dm, nullptr, Qbf, Dm, Bsz * Lt, Dm, Dm);
    gemm(KVbf, Dm, wk, Dm, bi + (l * 3 + 1) * Dm, nullptr, Kbf, Dm, Bsz * Gg, Dm, Dm);
    gemm(KVbf, Dm, wv, Dm, bi + (l * 3 + 2) * Dm, nullptr, Vbf, Dm, Bsz * Gg, Dm, Dm);
    attn_kernel<<<Bsz * Hh, 128, 0, stream>>>(Qbf, Kbf, Vbf, nullptr, Obf, Lt, Gg);
    gemm(Obf, Dm, woB + (long)l * Dm * Dm, Dm, bo + l * Dm, Hf, nullptr, Dm,
         Bsz * Lt, Dm, Dm);
    resln_kernel<<<Bsz * Lt, 256, 0, stream>>>(Af, Xbf, Hf, lng + l * Dm, lnb + l * Dm);
  }
  extract_t0_kernel<<<(Bsz * Dm + 255) / 256, 256, 0, stream>>>(Af, ztF);
  l2n_kernel<<<Bsz, 256, 0, stream>>>(ztF, ztN);

  // ---- g path: graph attends to text (masked keys) ----
  copyconv_kernel<<<8192, 256, 0, stream>>>(gf, Af, Xbf, (long)Bsz * Gg * Dm);
  copyconv_kernel<<<8192, 256, 0, stream>>>(tf, nullptr, KVbf, (long)Bsz * Lt * Dm);
  for (int l = 0; l < 2; ++l) {
    const unsigned short* wq = wiB + (long)(l * 3 + 0) * Dm * Dm;
    const unsigned short* wk = wiB + (long)(l * 3 + 1) * Dm * Dm;
    const unsigned short* wv = wiB + (long)(l * 3 + 2) * Dm * Dm;
    gemm(Xbf, Dm, wq, Dm, bi + (l * 3 + 0) * Dm, nullptr, Qbf, Dm, Bsz * Gg, Dm, Dm);
    gemm(KVbf, Dm, wk, Dm, bi + (l * 3 + 1) * Dm, nullptr, Kbf, Dm, Bsz * Lt, Dm, Dm);
    gemm(KVbf, Dm, wv, Dm, bi + (l * 3 + 2) * Dm, nullptr, Vbf, Dm, Bsz * Lt, Dm, Dm);
    attn_kernel<<<Bsz * Hh, 128, 0, stream>>>(Qbf, Kbf, Vbf, msk, Obf, Gg, Lt);
    gemm(Obf, Dm, woB + (long)l * Dm * Dm, Dm, bo + l * Dm, Hf, nullptr, Dm,
         Bsz * Gg, Dm, Dm);
    resln_kernel<<<Bsz * Gg, 256, 0, stream>>>(Af, Xbf, Hf, lng + l * Dm, lnb + l * Dm);
  }
  meang_kernel<<<(Bsz * Dm + 255) / 256, 256, 0, stream>>>(Af, zgF);
  l2n_kernel<<<Bsz, 256, 0, stream>>>(zgF, zgN);

  // ---- specific projections pt / pg ----
  gemm(zB, 1024, ptwB, S0n, ptb, ptF, nullptr, Dm, Bsz, Dm, S0n);
  gemm(zB + S0n, 1024, pgwB, S1n, pgb, pgF, nullptr, Dm, Bsz, Dm, S1n);
  l2n_kernel<<<Bsz, 256, 0, stream>>>(ptF, ptN);
  l2n_kernel<<<Bsz, 256, 0, stream>>>(pgF, pgN);

  // ---- four VQ passes (scores reuse Af; normalized inputs -> argmax of dot) ----
  gemm(ztN, Dm, cbn, Dm, nullptr, Af, nullptr, NEcb, Bsz, NEcb, Dm);
  vq_topk_kernel<<<Bsz, 256, 0, stream>>>(Af, NEcb, cbn, out, 0);

  gemm(zgN, Dm, cbn, Dm, nullptr, Af, nullptr, NEcb, Bsz, NEcb, Dm);
  vq_topk_kernel<<<Bsz, 256, 0, stream>>>(Af, NEcb, cbn, out, 512);

  gemm(ptN, Dm, cbn, Dm, nullptr, Af, nullptr, REGION, Bsz, REGION, Dm);
  vq_topk_kernel<<<Bsz, 256, 0, stream>>>(Af, REGION, cbn, out, 1024);

  const unsigned short* cbn2 = cbn + (long)(NEcb - REGION) * Dm;
  gemm(pgN, Dm, cbn2, Dm, nullptr, Af, nullptr, REGION, Bsz, REGION, Dm);
  vq_topk_kernel<<<Bsz, 256, 0, stream>>>(Af, REGION, cbn2, out, 1536);
}
